// GwACNode_28123445854595
// MI455X (gfx1250) — matrix operand order repaired and checked by reading, not compiled, not run on hardware
//
#include <hip/hip_runtime.h>
#include <stdint.h>
#include <math.h>

// Problem constants (from reference)
#define NNODES 512
#define FIN    32
#define HDIM   128
#define MDIM   64
#define ODIM   16
#define TSTEPS 512
#define KDIM   192      // H + M
#define BATCH  16       // starts per workgroup (WMMA M dim)
#define NBATCH 32       // 512 / 16
#define NT_S   8        // HDIM/16 state N-tiles
#define NT_M   4        // MDIM/16 msg N-tiles
#define KKN    6        // KDIM/32 k-steps per WMMA chain

typedef __attribute__((ext_vector_type(16))) _Float16 v16h;
typedef __attribute__((ext_vector_type(8)))  _Float16 h8;
typedef __attribute__((ext_vector_type(8)))  float    v8f;

// A-matrix fragment (16-bit, 16x32): lane L holds row M=L&15;
// halves 0..7 at K = k0..k0+7, halves 8..15 at K = k0+16..k0+23,
// where k0 = 32*kk + (L>=16 ? 8 : 0).  -> two 16B LDS loads.
__device__ inline v16h load_a_frag(const _Float16* p) {
  union { v16h v; h8 h[2]; } u;
  u.h[0] = *(const h8*)(p);
  u.h[1] = *(const h8*)(p + 16);
  return u.v;
}

// ---------------- encoder: enc16[n][h] = f16(x[n]@W_enc[h] + b_enc[h]) -----
__global__ void __launch_bounds__(HDIM)
enc_kernel(const float* __restrict__ x, const float* __restrict__ W_enc,
           const float* __restrict__ b_enc, _Float16* __restrict__ enc16) {
  int n = blockIdx.x;
  int h = threadIdx.x;
  float acc = b_enc[h];
#pragma unroll
  for (int f = 0; f < FIN; ++f) acc += x[n * FIN + f] * W_enc[h * FIN + f];
  enc16[n * HDIM + h] = (_Float16)acc;
}

// ---------------- replicate enc into per-start state arrays ----------------
__global__ void __launch_bounds__(256)
rep_kernel(const uint32_t* __restrict__ enc_dw, uint32_t* __restrict__ states_dw) {
  uint32_t idx = blockIdx.x * 256u + threadIdx.x;   // over 512*512*64 dwords
  states_dw[idx] = enc_dw[idx & (NNODES * HDIM / 2 - 1)];
}

// ---------------- main rollout: 32 WGs x 128 threads (4 waves) -------------
__global__ void __launch_bounds__(128)
rollout_kernel(const float* __restrict__ W_state, const float* __restrict__ b_state,
               const float* __restrict__ W_msg,  const float* __restrict__ b_msg,
               const int* __restrict__ popped,   const int* __restrict__ msg_src,
               _Float16* __restrict__ states,    _Float16* __restrict__ msgs) {
  // Staging for one-time weight swizzle: [ntile][kk][lane][16 halfs]
  __shared__ __attribute__((aligned(32))) _Float16 Bst[NT_S * KKN * 32 * 16]; // 48 KB
  __shared__ __attribute__((aligned(32))) _Float16 Bmg[NT_M * KKN * 32 * 16]; // 24 KB
  __shared__ __attribute__((aligned(32))) _Float16 Abuf[BATCH * KDIM];        // 6 KB [feat||msg]
  __shared__ __attribute__((aligned(32))) _Float16 NSbuf[BATCH * KDIM];       // 6 KB [ns||msg]
  __shared__ float bs_lds[HDIM];
  __shared__ float bm_lds[MDIM];
  __shared__ int   node_lds[2][BATCH];   // double-buffered schedule
  __shared__ int   src_lds[2][BATCH];

  const int tid    = threadIdx.x;
  const int lane   = tid & 31;
  const int wid    = tid >> 5;
  const int s_base = blockIdx.x * BATCH;

  // ---- one-time: pack weights into WMMA B-fragment order (f16) ----
  // B 32x16 f16 layout: lane n: col N=n&15; halves 0..7 at K=kb..kb+7, 8..15 at K=kb+16..kb+23 (kb=0 for n<16, 8 for n>=16).
  for (int idx = tid; idx < NT_S * KKN * 32 * 16; idx += 128) {
    int h  = idx & 15;
    int ln = (idx >> 4) & 31;
    int v  = idx >> 9;
    int kk = v % KKN, nt = v / KKN;
    int Kg = 32 * kk + ((ln >> 4) << 3) + (h & 7) + ((h >> 3) << 4);
    int Ng = 16 * nt + (ln & 15);
    Bst[idx] = (_Float16)W_state[Ng * KDIM + Kg];
  }
  for (int idx = tid; idx < NT_M * KKN * 32 * 16; idx += 128) {
    int h  = idx & 15;
    int ln = (idx >> 4) & 31;
    int v  = idx >> 9;
    int kk = v % KKN, nt = v / KKN;
    int Kg = 32 * kk + ((ln >> 4) << 3) + (h & 7) + ((h >> 3) << 4);
    int Ng = 16 * nt + (ln & 15);
    Bmg[idx] = (_Float16)W_msg[Ng * KDIM + Kg];
  }
  if (tid < HDIM) bs_lds[tid] = b_state[tid];
  if (tid < MDIM) bm_lds[tid] = b_msg[tid];
  // preload schedule for t=0 into buffer 0
  if (tid < BATCH) {
    node_lds[0][tid] = popped[(s_base + tid) * TSTEPS];
    src_lds[0][tid]  = msg_src[(s_base + tid) * TSTEPS];
  }
  __syncthreads();

  // ---- pin this wave's 18 B-fragments + biases in registers for all 512 steps
  v16h bfs0[KKN], bfs1[KKN], bfm[KKN];
#pragma unroll
  for (int kk = 0; kk < KKN; ++kk) {
    bfs0[kk] = *(const v16h*)&Bst[(((2 * wid) * KKN + kk) * 32 + lane) * 16];
    bfs1[kk] = *(const v16h*)&Bst[(((2 * wid + 1) * KKN + kk) * 32 + lane) * 16];
    bfm[kk]  = *(const v16h*)&Bmg[((wid * KKN + kk) * 32 + lane) * 16];
  }
  const int   Nl      = lane & 15;
  const int   Mb      = (lane >> 4) * 8;
  const float bias_s0 = bs_lds[(2 * wid) * 16 + Nl];
  const float bias_s1 = bs_lds[(2 * wid + 1) * 16 + Nl];
  const float bias_m  = bm_lds[wid * 16 + Nl];
  const int   ar      = (lane & 15) * KDIM + ((lane >> 4) ? 8 : 0);

  uint32_t*       A_dw        = (uint32_t*)Abuf;
  uint32_t*       NS_dw       = (uint32_t*)NSbuf;
  const uint32_t* states_dw_c = (const uint32_t*)states;
  uint32_t*       states_dw   = (uint32_t*)states;
  const uint32_t* msgs_dw     = (const uint32_t*)msgs;

#pragma unroll 1
  for (int t = 0; t < TSTEPS; ++t) {
    const int cur = t & 1;
    const int nxt = cur ^ 1;

    // gather feat: 16 rows x 64 dwords from states (L2-resident)
    {
      int i = tid >> 3, c = tid & 7;
      const uint32_t* sp =
          states_dw_c + ((size_t)(s_base + i) * NNODES + node_lds[cur][i]) * (HDIM / 2) + c * 8;
      uint32_t* dp = A_dw + i * (KDIM / 2) + c * 8;
#pragma unroll
      for (int q = 0; q < 8; ++q) dp[q] = sp[q];
    }
    // gather msg: 16 rows x 32 dwords (ones if src<0); mirror into NSbuf
    if (tid < 64) {
      int i = tid >> 2, c = tid & 3;
      int sc = src_lds[cur][i];
      uint32_t vbuf[8];
      if (sc < 0) {
#pragma unroll
        for (int q = 0; q < 8; ++q) vbuf[q] = 0x3C003C00u;  // packed f16 (1.0,1.0)
      } else {
        const uint32_t* sp =
            msgs_dw + ((size_t)(s_base + i) * TSTEPS + sc) * (MDIM / 2) + c * 8;
#pragma unroll
        for (int q = 0; q < 8; ++q) vbuf[q] = sp[q];
      }
      uint32_t* d0 = A_dw  + i * (KDIM / 2) + (HDIM / 2) + c * 8;
      uint32_t* d1 = NS_dw + i * (KDIM / 2) + (HDIM / 2) + c * 8;
#pragma unroll
      for (int q = 0; q < 8; ++q) { d0[q] = vbuf[q]; d1[q] = vbuf[q]; }
    }
    __syncthreads();  // A ready

    // prefetch next step's schedule while GEMMs run (consumed after next barrier pair)
    if (tid < BATCH && t + 1 < TSTEPS) {
      int g = (s_base + tid) * TSTEPS + t + 1;
      node_lds[nxt][tid] = popped[g];
      src_lds[nxt][tid]  = msg_src[g];
    }

    // ---- state GEMM: [16,192]x[192,128], relu, write ns cols of NSbuf ----
    {
      v8f acc0 = {0.f, 0.f, 0.f, 0.f, 0.f, 0.f, 0.f, 0.f};
      v8f acc1 = {0.f, 0.f, 0.f, 0.f, 0.f, 0.f, 0.f, 0.f};
#pragma unroll
      for (int kk = 0; kk < KKN; ++kk) {
        v16h a = load_a_frag(&Abuf[ar + 32 * kk]);
        acc0 = __builtin_amdgcn_wmma_f32_16x16x32_f16(false, a, false, bfs0[kk],
                                                      (short)0, acc0, false, false);
        acc1 = __builtin_amdgcn_wmma_f32_16x16x32_f16(false, a, false, bfs1[kk],
                                                      (short)0, acc1, false, false);
      }
#pragma unroll
      for (int r = 0; r < 8; ++r) {
        float v0 = acc0[r] + bias_s0;
        float v1 = acc1[r] + bias_s1;
        v0 = v0 > 0.f ? v0 : 0.f;
        v1 = v1 > 0.f ? v1 : 0.f;
        NSbuf[(Mb + r) * KDIM + (2 * wid) * 16 + Nl]     = (_Float16)v0;
        NSbuf[(Mb + r) * KDIM + (2 * wid + 1) * 16 + Nl] = (_Float16)v1;
      }
    }
    __syncthreads();  // ns ready

    // ---- msg GEMM: [16,192]x[192,64] -> msgs[s][t][:] ----
    {
      v8f acc = {0.f, 0.f, 0.f, 0.f, 0.f, 0.f, 0.f, 0.f};
#pragma unroll
      for (int kk = 0; kk < KKN; ++kk) {
        v16h a = load_a_frag(&NSbuf[ar + 32 * kk]);
        acc = __builtin_amdgcn_wmma_f32_16x16x32_f16(false, a, false, bfm[kk],
                                                     (short)0, acc, false, false);
      }
#pragma unroll
      for (int r = 0; r < 8; ++r) {
        int i = Mb + r;
        msgs[((size_t)(s_base + i) * TSTEPS + t) * MDIM + wid * 16 + Nl] =
            (_Float16)(acc[r] + bias_m);
      }
    }

    // ---- scatter ns back to states[s][node] ----
    {
      int i = tid >> 3, c = tid & 7;
      const uint32_t* sp = NS_dw + i * (KDIM / 2) + c * 8;
      uint32_t* dp =
          states_dw + ((size_t)(s_base + i) * NNODES + node_lds[cur][i]) * (HDIM / 2) + c * 8;
#pragma unroll
      for (int q = 0; q < 8; ++q) dp[q] = sp[q];
    }
    __threadfence_block();  // global msg/state writes visible to later gathers
    __syncthreads();        // end of step: Abuf/NSbuf consumers done, sched[nxt] visible
  }
}

// ---------------- decoder + log_softmax on the diagonal --------------------
__global__ void __launch_bounds__(128)
dec_kernel(const _Float16* __restrict__ states, const float* __restrict__ W_dec,
           const float* __restrict__ b_dec, float* __restrict__ out) {
  __shared__ float lg[8][ODIM];
  int tid = threadIdx.x;
  int rl  = tid >> 4;
  int o   = tid & 15;
  int row = blockIdx.x * 8 + rl;
  const _Float16* sp = states + ((size_t)row * NNODES + row) * HDIM;
  float acc = b_dec[o];
#pragma unroll 4
  for (int h = 0; h < HDIM; ++h) acc += (float)sp[h] * W_dec[o * HDIM + h];
  lg[rl][o] = acc;
  __syncthreads();
  float m = lg[rl][0];
#pragma unroll
  for (int j = 1; j < ODIM; ++j) m = fmaxf(m, lg[rl][j]);
  float s = 0.f;
#pragma unroll
  for (int j = 0; j < ODIM; ++j) s += expf(lg[rl][j] - m);
  out[row * ODIM + o] = acc - m - logf(s);
}

extern "C" void kernel_launch(void* const* d_in, const int* in_sizes, int n_in,
                              void* d_out, int out_size, void* d_ws, size_t ws_size,
                              hipStream_t stream) {
  const float* x       = (const float*)d_in[0];
  const float* W_enc   = (const float*)d_in[1];
  const float* b_enc   = (const float*)d_in[2];
  const float* W_state = (const float*)d_in[3];
  const float* b_state = (const float*)d_in[4];
  const float* W_msg   = (const float*)d_in[5];
  const float* b_msg   = (const float*)d_in[6];
  const float* W_dec   = (const float*)d_in[7];
  const float* b_dec   = (const float*)d_in[8];
  const int*   popped  = (const int*)d_in[9];
  const int*   msg_src = (const int*)d_in[10];
  float*       out     = (float*)d_out;

  // workspace layout (all f16): states 64MB | msgs 32MB | enc 128KB  (~96MB, L2-resident)
  char* ws = (char*)d_ws;
  _Float16* states = (_Float16*)(ws);
  _Float16* msgs   = (_Float16*)(ws + (size_t)NNODES * NNODES * HDIM * 2);
  _Float16* enc16  = (_Float16*)(ws + (size_t)NNODES * NNODES * HDIM * 2 +
                                      (size_t)NNODES * TSTEPS * MDIM * 2);

  enc_kernel<<<NNODES, HDIM, 0, stream>>>(x, W_enc, b_enc, enc16);
  rep_kernel<<<(NNODES * NNODES * (HDIM / 2)) / 256, 256, 0, stream>>>(
      (const uint32_t*)enc16, (uint32_t*)states);
  rollout_kernel<<<NBATCH, 128, 0, stream>>>(W_state, b_state, W_msg, b_msg,
                                             popped, msg_src, states, msgs);
  dec_kernel<<<NNODES / 8, 128, 0, stream>>>(states, W_dec, b_dec, out);
}
